// MultiLayerGRU_91036126806106
// MI455X (gfx1250) — matrix operand order repaired and checked by reading, not compiled, not run on hardware
//
#include <hip/hip_runtime.h>
#include <hip/hip_bf16.h>

// ---------------------------------------------------------------------------
// 2-layer GRU + linear head for MI455X (gfx1250, wave32, WMMA).
//
// ~378 GFLOP total; the 1024 sequential recurrent steps dominate latency.
// Per step, the 6 MB bf16 Whh stays L2-resident (192 MB L2). We fuse
// layer0[t] + layer1[t-1] into one kernel -> 513 launches on the critical
// path. All matmuls use v_wmma_f32_16x16x32_bf16 (bf16 in, f32 accumulate).
//
// This revision:
//  * A panels (h, x_t) staged to LDS once per block via the CDNA5 async
//    global->LDS path (global_load_async_to_lds_b128 + s_wait_asynccnt),
//    killing the 6x duplicate A fragment loads across N-tile waves.
//  * B (weight) fragments double-buffered so each WMMA waits only on
//    older loads instead of s_wait_loadcnt 0 every step.
// ---------------------------------------------------------------------------

#define B_    32
#define S_    512
#define D_IN  512
#define H_    1024
#define G3    (3 * H_)

typedef __attribute__((ext_vector_type(16))) __bf16 v16bf;
typedef __attribute__((ext_vector_type(8)))  float  v8f;

union FragBF {
  v16bf bf;
  uint4 q[2];
};

__device__ __forceinline__ unsigned short f32_to_bf16_rne(float f) {
  unsigned int u = __float_as_uint(f);
  unsigned int r = u + 0x7FFFu + ((u >> 16) & 1u);
  return (unsigned short)(r >> 16);
}

// CDNA5 async global->LDS copy, 16B per lane. VDST VGPR carries the LDS byte
// offset (hw adds LDS_BASE); tracked by ASYNCcnt.
__device__ __forceinline__ void async_g2l_b128(unsigned ldsOff, const void* g) {
  asm volatile("global_load_async_to_lds_b128 %0, %1, off"
               :: "v"(ldsOff), "v"((unsigned long long)(size_t)g) : "memory");
}
__device__ __forceinline__ void wait_async0() {
  asm volatile("s_wait_asynccnt 0" ::: "memory");
}

// A-matrix (16x32 bf16, row-major source). ISA layout:
// lanes 0-15: row M=lane, elems 0..7 = K 0..7,  elems 8..15 = K 16..23
// lanes 16-31: row M=lane-16, elems 0..7 = K 8..15, elems 8..15 = K 24..31
__device__ __forceinline__ v16bf load_a_frag(const unsigned short* tileBase,
                                             long rowStride, int kbase, int lane) {
  const int m  = lane & 15;
  const int hi = (lane >> 4) * 8;
  const unsigned short* p = tileBase + (long)m * rowStride + kbase + hi;
  FragBF f;
  f.q[0] = *reinterpret_cast<const uint4*>(p);        // K = kbase+hi .. +7
  f.q[1] = *reinterpret_cast<const uint4*>(p + 16);   // K = kbase+16+hi .. +7
  return f.bf;
}

// B-matrix (32x16 bf16). Source is row-major weight W[outRow][k] (== B^T), so
// each lane reads K = kbase+hi..+7 and kbase+16+hi..+7 of its column (hi = 8*(lane>>4)).
__device__ __forceinline__ FragBF load_b_frag(const unsigned short* W, long ldk,
                                              int outRowBase, int kbase, int lane) {
  const int n  = lane & 15;
  const int k2 = (lane >> 4) * 8;
  const unsigned short* p = W + (long)(outRowBase + n) * ldk + kbase + k2;
  FragBF f;
  f.q[0] = *reinterpret_cast<const uint4*>(p);
  f.q[1] = *reinterpret_cast<const uint4*>(p + 16);
  return f;
}

// --------------------------- helper kernels --------------------------------

__global__ void cvt_f32_bf16(const float* __restrict__ src,
                             unsigned short* __restrict__ dst, int n) {
  int i = blockIdx.x * blockDim.x + threadIdx.x;
  int stride = gridDim.x * blockDim.x;
  for (; i < n; i += stride) dst[i] = f32_to_bf16_rne(src[i]);
}

__global__ void zero_u32(unsigned int* __restrict__ p, int n) {
  int i = blockIdx.x * blockDim.x + threadIdx.x;
  int stride = gridDim.x * blockDim.x;
  for (; i < n; i += stride) p[i] = 0u;
}

__global__ void copy_hidden(const float* __restrict__ h0,
                            const float* __restrict__ h1,
                            float* __restrict__ outh) {
  int i = blockIdx.x * blockDim.x + threadIdx.x;   // 65536 threads
  if (i < B_ * H_)          outh[i] = h0[i];
  else if (i < 2 * B_ * H_) outh[i] = h1[i - B_ * H_];
}

// --------------------------- fused GRU step --------------------------------
// Grid = 64 blocks: blocks 0..31 -> layer0 step t, blocks 32..63 -> layer1
// step t-1. Each block (12 waves) owns 32 hidden units: computes r/z/n gate
// tiles (2 M x 6 N of 16x16, K = 1024 hidden + Kx input), exchanges via LDS,
// then applies:
//   r = sig(xr+hr+br), z = sig(xz+hz+bz), n = tanh(xn+bn+(1+r)*hn)   [Whn==Whh_n]
//   h' = (1-z)*n + z*h
__global__ __launch_bounds__(384) void gru_step_kernel(
    int t,
    const unsigned short* __restrict__ xbf,                 // [B][S][D_IN] bf16
    const unsigned short* __restrict__ Wih0b, const unsigned short* __restrict__ Whh0b,
    const float* __restrict__ b0,
    const unsigned short* __restrict__ Wih1b, const unsigned short* __restrict__ Whh1b,
    const float* __restrict__ b1,
    unsigned short* y0bf,                                   // [S][B][H] bf16 (r/w)
    unsigned short* __restrict__ y1bf,                      // [S][B][H] bf16
    float* h0f, unsigned short* h0bf,                       // [2][B][H] ping-pong
    float* h1f, unsigned short* h1bf) {
  const int layer = (int)blockIdx.x >> 5;
  const int cb    = ((int)blockIdx.x & 31) * 32;            // hidden-unit base

  int tcur, Kx;
  long xstride;
  const unsigned short *xin, *Wih, *Whh, *hbf_r;
  const float *bias, *hf_r;
  float *hf_w;
  unsigned short *hbf_w, *ybf;

  if (layer == 0) {
    if (t >= S_) return;
    tcur = t;
    xin = xbf + (long)t * D_IN;  xstride = (long)S_ * D_IN;  Kx = D_IN;
    Wih = Wih0b; Whh = Whh0b; bias = b0;
    const int pr = t & 1, pw = pr ^ 1;
    hf_r  = h0f  + pr * (B_ * H_);  hf_w  = h0f  + pw * (B_ * H_);
    hbf_r = h0bf + pr * (B_ * H_);  hbf_w = h0bf + pw * (B_ * H_);
    ybf = y0bf;
  } else {
    if (t == 0) return;
    tcur = t - 1;
    xin = y0bf + (long)tcur * (B_ * H_);  xstride = H_;  Kx = H_;
    Wih = Wih1b; Whh = Whh1b; bias = b1;
    const int pr = tcur & 1, pw = pr ^ 1;
    hf_r  = h1f  + pr * (B_ * H_);  hf_w  = h1f  + pw * (B_ * H_);
    hbf_r = h1bf + pr * (B_ * H_);  hbf_w = h1bf + pw * (B_ * H_);
    ybf = y1bf;
  }

  const int lane = threadIdx.x & 31;
  const int w    = threadIdx.x >> 5;   // 0..11
  const int mi   = w & 1;              // M tile
  const int nt   = w >> 1;             // 0..5
  const int g    = nt >> 1;            // gate: 0=r 1=z 2=n
  const int nj   = nt & 1;             // N subtile within the 32-unit slice
  const int m0   = mi * 16;
  const int colBase = g * H_ + cb + nj * 16;   // row of W == output column

  // LDS: A panels (async-staged) + gate exchange. 64 + 64 + 16 KB = 144 KB.
  __shared__ __align__(16) unsigned short s_h[B_ * H_];   // h panel [32][H_]
  __shared__ __align__(16) unsigned short s_x[B_ * H_];   // x panel [32][Kx]
  __shared__ float s_r [B_][32];
  __shared__ float s_z [B_][32];
  __shared__ float s_xn[B_][32];
  __shared__ float s_hn[B_][32];

  // ---- async global->LDS staging of the A panels (16B chunks per lane) ----
  {
    const unsigned ldsH = (unsigned)(size_t)&s_h[0];
    const int chunksH = (B_ * H_) / 8;                 // 4096 x 16B
    for (int i = threadIdx.x; i < chunksH; i += 384)
      async_g2l_b128(ldsH + (unsigned)i * 16u, hbf_r + (long)i * 8);

    const unsigned ldsX = (unsigned)(size_t)&s_x[0];
    const int cpr = Kx / 8;                            // chunks per row
    for (int i = threadIdx.x; i < B_ * cpr; i += 384) {
      const int row = i / cpr, c = i - row * cpr;
      async_g2l_b128(ldsX + (unsigned)(row * Kx + c * 8) * 2u,
                     xin + (long)row * xstride + c * 8);
    }
    wait_async0();        // each wave drains its own ASYNCcnt
    __syncthreads();      // then all LDS writes are visible block-wide
  }

  v8f acc_h = {};
  v8f acc_x = {};

  // hidden part: h(32xH) @ Whh[colBase..+16]^T ; A from LDS, B double-buffered
  {
    const unsigned short* aBase = &s_h[m0 * H_];
    FragBF bcur = load_b_frag(Whh, H_, colBase, 0, lane);
    for (int kb = 0; kb < H_ - 32; kb += 32) {
      FragBF bnxt = load_b_frag(Whh, H_, colBase, kb + 32, lane);
      v16bf a = load_a_frag(aBase, H_, kb, lane);      // ds_load_b128 x2
      acc_h = __builtin_amdgcn_wmma_f32_16x16x32_bf16(false, a, false, bcur.bf,
                                                      (short)0, acc_h, false, false);
      bcur = bnxt;
    }
    v16bf a = load_a_frag(aBase, H_, H_ - 32, lane);
    acc_h = __builtin_amdgcn_wmma_f32_16x16x32_bf16(false, a, false, bcur.bf,
                                                    (short)0, acc_h, false, false);
  }
  // input part: x_t(32xKx) @ Wih[colBase..+16]^T
  {
    const unsigned short* aBase = &s_x[m0 * Kx];
    FragBF bcur = load_b_frag(Wih, Kx, colBase, 0, lane);
    for (int kb = 0; kb < Kx - 32; kb += 32) {
      FragBF bnxt = load_b_frag(Wih, Kx, colBase, kb + 32, lane);
      v16bf a = load_a_frag(aBase, Kx, kb, lane);
      acc_x = __builtin_amdgcn_wmma_f32_16x16x32_bf16(false, a, false, bcur.bf,
                                                      (short)0, acc_x, false, false);
      bcur = bnxt;
    }
    v16bf a = load_a_frag(aBase, Kx, Kx - 32, lane);
    acc_x = __builtin_amdgcn_wmma_f32_16x16x32_bf16(false, a, false, bcur.bf,
                                                    (short)0, acc_x, false, false);
  }

  // C/D layout: elem e -> M = e + (lane>>4)*8, N = lane&15
  {
    const int nl = lane & 15;
    const int mo = (lane >> 4) * 8;
    const int j  = nj * 16 + nl;
#pragma unroll
    for (int e = 0; e < 8; ++e) {
      const int m = m0 + mo + e;
      if (g == 0)      s_r[m][j] = acc_h[e] + acc_x[e];
      else if (g == 1) s_z[m][j] = acc_h[e] + acc_x[e];
      else { s_xn[m][j] = acc_x[e]; s_hn[m][j] = acc_h[e]; }
    }
  }
  __syncthreads();

  // pointwise GRU update for this block's 32 units x 32 batches
  for (int idx = threadIdx.x; idx < B_ * 32; idx += blockDim.x) {
    const int m = idx >> 5, j = idx & 31;
    const int unit = cb + j;
    const float rv = 1.f / (1.f + __expf(-(s_r[m][j] + bias[unit])));
    const float zv = 1.f / (1.f + __expf(-(s_z[m][j] + bias[H_ + unit])));
    const float nv = tanhf(s_xn[m][j] + bias[2 * H_ + unit] + (1.f + rv) * s_hn[m][j]);
    const float hold = hf_r[m * H_ + unit];
    const float hnew = (1.f - zv) * nv + zv * hold;
    hf_w[m * H_ + unit] = hnew;
    const unsigned short hb = f32_to_bf16_rne(hnew);
    hbf_w[m * H_ + unit] = hb;
    ybf[(long)tcur * (B_ * H_) + m * H_ + unit] = hb;
  }
}

// --------------------------- output projection -----------------------------
// out[b,t,:] = y1[t,b,:] @ Wlin^T + blin. Rows r = t*32+b (y1 layout [S][B][H]).
// 8 waves/block, block tile 32(M) x 64(N), K=1024; A and B double-buffered.
__global__ __launch_bounds__(256) void linear_out_kernel(
    const unsigned short* __restrict__ y1bf,
    const unsigned short* __restrict__ Wlinb,
    const float* __restrict__ blin,
    float* __restrict__ out) {
  const int lane = threadIdx.x & 31;
  const int w    = threadIdx.x >> 5;          // 0..7
  const int mi   = w & 1;
  const int ni   = w >> 1;                    // 0..3
  const int bm   = (int)blockIdx.x & 511;     // 512 M blocks
  const int bn   = (int)blockIdx.x >> 9;      // 8  N blocks
  const int rows    = bm * 32 + mi * 16;
  const int colBase = bn * 64 + ni * 16;

  v8f acc = {};
  const unsigned short* aP = y1bf + (long)rows * H_;

  FragBF acur; acur.bf = load_a_frag(aP, H_, 0, lane);
  FragBF bcur = load_b_frag(Wlinb, H_, colBase, 0, lane);
  for (int kb = 0; kb < H_ - 32; kb += 32) {
    FragBF anxt; anxt.bf = load_a_frag(aP, H_, kb + 32, lane);
    FragBF bnxt = load_b_frag(Wlinb, H_, colBase, kb + 32, lane);
    acc = __builtin_amdgcn_wmma_f32_16x16x32_bf16(false, acur.bf, false, bcur.bf,
                                                  (short)0, acc, false, false);
    acur = anxt; bcur = bnxt;
  }
  acc = __builtin_amdgcn_wmma_f32_16x16x32_bf16(false, acur.bf, false, bcur.bf,
                                                (short)0, acc, false, false);

  const int nl  = lane & 15;
  const int mo  = (lane >> 4) * 8;
  const int col = colBase + nl;
  const float bv = blin[col];
#pragma unroll
  for (int e = 0; e < 8; ++e) {
    const int grow = rows + mo + e;           // = t*32 + b
    const int tt = grow >> 5, bb = grow & 31;
    out[(long)bb * (S_ * D_IN) + (long)tt * D_IN + col] = acc[e] + bv;
  }
}

// ------------------------------- launcher ----------------------------------

extern "C" void kernel_launch(void* const* d_in, const int* in_sizes, int n_in,
                              void* d_out, int out_size, void* d_ws, size_t ws_size,
                              hipStream_t stream) {
  (void)in_sizes; (void)n_in; (void)out_size; (void)ws_size;
  const float* x    = (const float*)d_in[0];
  const float* Wih0 = (const float*)d_in[1];
  const float* Whh0 = (const float*)d_in[2];
  const float* b0   = (const float*)d_in[3];
  const float* Wih1 = (const float*)d_in[4];
  const float* Whh1 = (const float*)d_in[5];
  const float* b1   = (const float*)d_in[6];
  const float* Wlin = (const float*)d_in[7];
  const float* blin = (const float*)d_in[8];
  float* out = (float*)d_out;

  // ---- workspace carve-up (~107 MB, all 256B-aligned chunks) ----
  char* p = (char*)d_ws;
  auto carve = [&](size_t bytes) { char* q = p; p += (bytes + 255) & ~(size_t)255; return q; };
  unsigned short* xbf   = (unsigned short*)carve((size_t)B_ * S_ * D_IN * 2);
  unsigned short* Wih0b = (unsigned short*)carve((size_t)G3 * D_IN * 2);
  unsigned short* Whh0b = (unsigned short*)carve((size_t)G3 * H_ * 2);
  unsigned short* Wih1b = (unsigned short*)carve((size_t)G3 * H_ * 2);
  unsigned short* Whh1b = (unsigned short*)carve((size_t)G3 * H_ * 2);
  unsigned short* Wlinb = (unsigned short*)carve((size_t)D_IN * H_ * 2);
  unsigned short* y0bf  = (unsigned short*)carve((size_t)S_ * B_ * H_ * 2);
  unsigned short* y1bf  = (unsigned short*)carve((size_t)S_ * B_ * H_ * 2);
  float*          h0f   = (float*)carve(2ull * B_ * H_ * 4);
  float*          h1f   = (float*)carve(2ull * B_ * H_ * 4);
  unsigned short* h0bf  = (unsigned short*)carve(2ull * B_ * H_ * 2);
  unsigned short* h1bf  = (unsigned short*)carve(2ull * B_ * H_ * 2);

  // ---- one-time conversions (fp32 -> bf16) ----
  cvt_f32_bf16<<<2048, 256, 0, stream>>>(x,    xbf,   B_ * S_ * D_IN);
  cvt_f32_bf16<<<512,  256, 0, stream>>>(Wih0, Wih0b, G3 * D_IN);
  cvt_f32_bf16<<<1024, 256, 0, stream>>>(Whh0, Whh0b, G3 * H_);
  cvt_f32_bf16<<<1024, 256, 0, stream>>>(Wih1, Wih1b, G3 * H_);
  cvt_f32_bf16<<<1024, 256, 0, stream>>>(Whh1, Whh1b, G3 * H_);
  cvt_f32_bf16<<<256,  256, 0, stream>>>(Wlin, Wlinb, D_IN * H_);

  // zero h ping-pong buffers (contiguous: h0f,h1f,h0bf,h1bf = 768 KB)
  zero_u32<<<192, 256, 0, stream>>>((unsigned int*)h0f,
                                    (int)((2ull * B_ * H_ * (4 + 4 + 2 + 2)) / 4));

  // ---- recurrent pipeline: kernel t does layer0[t] and layer1[t-1] ----
  for (int t = 0; t <= S_; ++t) {
    gru_step_kernel<<<64, 384, 0, stream>>>(t, xbf, Wih0b, Whh0b, b0,
                                            Wih1b, Whh1b, b1,
                                            y0bf, y1bf, h0f, h0bf, h1f, h1bf);
  }

  // ---- output projection + final hidden states ----
  linear_out_kernel<<<4096, 256, 0, stream>>>(y1bf, Wlinb, blin, out);
  // S_ even -> final h lives in parity 0 of each ping-pong buffer
  copy_hidden<<<256, 256, 0, stream>>>(h0f, h1f, out + (size_t)B_ * S_ * D_IN);
}
